// TopologyStream_73151882985679
// MI455X (gfx1250) — matrix operand order repaired and checked by reading, not compiled, not run on hardware
//
#include <hip/hip_runtime.h>
#include <hip/hip_bf16.h>

typedef __attribute__((ext_vector_type(16))) _Float16 v16h;
typedef __attribute__((ext_vector_type(8)))  _Float16 v8h;
typedef __attribute__((ext_vector_type(8)))  float    v8f;

#define NEG_SLOPE 0.2f

// ---------- small utility kernels ----------
__global__ void k_fill_f32(float* p, float v, int n) {
    int t = blockIdx.x * blockDim.x + threadIdx.x;
    if (t < n) p[t] = v;
}
__global__ void k_fill_u32(unsigned* p, unsigned v, int n) {
    int t = blockIdx.x * blockDim.x + threadIdx.x;
    if (t < n) p[t] = v;
}
__global__ void k_f32_to_f16(const float* __restrict__ x, _Float16* __restrict__ y, int n) {
    int t = blockIdx.x * blockDim.x + threadIdx.x;
    if (t < n) y[t] = (_Float16)x[t];
}
// W [K, Ncols] row-major f32  ->  Wt [Ncols, K] row-major f16 (column-major view of W)
__global__ void k_transpose_to_f16(const float* __restrict__ W, _Float16* __restrict__ Wt,
                                   int K, int Ncols) {
    int t = blockIdx.x * blockDim.x + threadIdx.x;
    if (t < K * Ncols) {
        int k = t / Ncols, n = t - k * Ncols;
        Wt[n * K + k] = (_Float16)W[t];
    }
}
// out[t] = bias[t & cmask]   (C is a power of two)
__global__ void k_bias_init(float* __restrict__ out, const float* __restrict__ bias,
                            int n, int cmask) {
    int t = blockIdx.x * blockDim.x + threadIdx.x;
    if (t < n) out[t] = bias[t & cmask];
}
// y = f16(elu(acc + bias))
__global__ void k_bias_elu_f16(const float* __restrict__ acc, const float* __restrict__ bias,
                               _Float16* __restrict__ y, int n, int cmask) {
    int t = blockIdx.x * blockDim.x + threadIdx.x;
    if (t < n) {
        float v = acc[t] + bias[t & cmask];
        v = v > 0.f ? v : expm1f(v);
        y[t] = (_Float16)v;
    }
}

// ---------- WMMA GEMM: C[M,Ncols] = A[M,128](f16) * B[128,Ncols], Bt = B^T (f16, row-major [Ncols,128])
// grid.x = M/16 ; blockDim.x = (Ncols/16)*32 ; one wave per 16x16 tile, K=128 in 4 WMMA steps.
__global__ __launch_bounds__(256) void k_wmma_gemm(const _Float16* __restrict__ A,
                                                   const _Float16* __restrict__ Bt,
                                                   float* __restrict__ C, int Ncols) {
    const int lane = threadIdx.x & 31;
    const int wave = threadIdx.x >> 5;        // n-tile index
    const int m0   = blockIdx.x * 16;
    const int row  = lane & 15;
    const int hi   = lane >> 4;               // 0: lanes 0-15, 1: lanes 16-31

    // A fragment: row = m0+row, halves = K in [hi*8, hi*8+8) and [hi*8+16, hi*8+24) per k-tile
    const _Float16* Ar = A  + (size_t)(m0 + row) * 128 + hi * 8;
    // B fragment: col = wave*16+row, halves = K in [hi*8, hi*8+8) and [hi*8+16, hi*8+24) per k-tile
    const _Float16* Br = Bt + (size_t)(wave * 16 + row) * 128 + hi * 8;

    v8f c = {};
#pragma unroll
    for (int kt = 0; kt < 4; ++kt) {
        v8h a0 = *(const v8h*)(Ar + kt * 32);
        v8h a1 = *(const v8h*)(Ar + kt * 32 + 16);
        v8h b0 = *(const v8h*)(Br + kt * 32);
        v8h b1 = *(const v8h*)(Br + kt * 32 + 16);
        v16h a = __builtin_shufflevector(a0, a1, 0,1,2,3,4,5,6,7,8,9,10,11,12,13,14,15);
        v16h b = __builtin_shufflevector(b0, b1, 0,1,2,3,4,5,6,7,8,9,10,11,12,13,14,15);
        c = __builtin_amdgcn_wmma_f32_16x16x32_f16(false, a, false, b, (short)0, c, false, false);
    }
    // C layout: VGPR r -> row m0 + hi*8 + r, col wave*16 + row
    float* Cp = C + (size_t)(m0 + hi * 8) * Ncols + wave * 16 + row;
#pragma unroll
    for (int r = 0; r < 8; ++r) Cp[(size_t)r * Ncols] = c[r];
}

// ---------- per-node attention coefficients: AS[n,h]=<H[n,h,:],a_src[h,:]> ----------
template <int H, int F>
__global__ void k_alphas(const float* __restrict__ Hf, const float* __restrict__ a_src,
                         const float* __restrict__ a_dst, float* __restrict__ AS,
                         float* __restrict__ AD, int N) {
    int t = blockIdx.x * blockDim.x + threadIdx.x;
    if (t >= N * H) return;
    const int h = (H == 1) ? 0 : (t & (H - 1));
    const float* hp = Hf + (size_t)t * F;
    const float* sp = a_src + h * F;
    const float* dp = a_dst + h * F;
    float s = 0.f, d = 0.f;
#pragma unroll
    for (int f = 0; f < F; f += 4) {
        float4 v = *(const float4*)(hp + f);
        float4 a = *(const float4*)(sp + f);
        float4 b = *(const float4*)(dp + f);
        s += v.x * a.x + v.y * a.y + v.z * a.z + v.w * a.w;
        d += v.x * b.x + v.y * b.y + v.z * b.z + v.w * b.w;
    }
    AS[t] = s; AD[t] = d;
}

// ---------- order-preserving float<->uint for atomic max ----------
__device__ __forceinline__ unsigned f2ord(float f) {
    unsigned u = __float_as_uint(f);
    return (u & 0x80000000u) ? ~u : (u | 0x80000000u);
}
__device__ __forceinline__ float ord2f(unsigned u) {
    unsigned b = (u & 0x80000000u) ? (u & 0x7FFFFFFFu) : ~u;
    return __uint_as_float(b);
}

// edges e in [0,E) come from edge_index; e in [E, EA) are self loops (s=d=e-E)
// Pass 1: compute leaky-relu logit, stash it in EX, atomic-max per (dst, head).
template <int H>
__global__ void k_edge_max(const int* __restrict__ src, const int* __restrict__ dst,
                           const float* __restrict__ AS, const float* __restrict__ AD,
                           unsigned* __restrict__ M, float* __restrict__ EX,
                           int E, int EA) {
    int t = blockIdx.x * blockDim.x + threadIdx.x;
    if (t >= EA * H) return;
    const int e = (H == 1) ? t : (t >> 1);
    const int h = (H == 1) ? 0 : (t & 1);
    int s = (e < E) ? src[e] : (e - E);
    int d = (e < E) ? dst[e] : (e - E);
    float v = AS[s * H + h] + AD[d * H + h];
    v = v > 0.f ? v : NEG_SLOPE * v;
    EX[t] = v;
    atomicMax(M + d * H + h, f2ord(v));
}

// Pass 2: EX holds the logit; replace with exp(logit - max), atomic-sum denominators.
template <int H>
__global__ void k_edge_exp(const int* __restrict__ dst, const unsigned* __restrict__ M,
                           float* __restrict__ EX, float* __restrict__ DEN,
                           int E, int EA) {
    int t = blockIdx.x * blockDim.x + threadIdx.x;
    if (t >= EA * H) return;
    const int e = (H == 1) ? t : (t >> 1);
    const int h = (H == 1) ? 0 : (t & 1);
    int d = (e < E) ? dst[e] : (e - E);
    float ex = expf(EX[t] - ord2f(M[d * H + h]));
    EX[t] = ex;
    atomicAdd(DEN + d * H + h, ex);
}

// Pass 3: one thread per (edge, 4 features): b128 gather + 4 atomic adds.
template <int H, int F>
__global__ void k_edge_scatter(const int* __restrict__ src, const int* __restrict__ dst,
                               const float* __restrict__ Hf, const float* __restrict__ EX,
                               const float* __restrict__ DEN, float* __restrict__ ACC,
                               int E, int EA) {
    constexpr int HF = H * F;
    constexpr int Q  = HF / 4;                 // quads per edge
    int t = blockIdx.x * blockDim.x + threadIdx.x;
    if (t >= EA * Q) return;
    const int e = t / Q;                       // Q is a compile-time pow2 -> shift
    const int q = t - e * Q;
    const int r = q * 4;
    const int h = (H == 1) ? 0 : (r >> 6);     // r / F with F=64
    int s = (e < E) ? src[e] : (e - E);
    int d = (e < E) ? dst[e] : (e - E);
    float coef = EX[e * H + h] / DEN[d * H + h];
    float4 v = *(const float4*)(Hf + (size_t)s * HF + r);
    float* a = ACC + (size_t)d * HF + r;
    atomicAdd(a + 0, coef * v.x);
    atomicAdd(a + 1, coef * v.y);
    atomicAdd(a + 2, coef * v.z);
    atomicAdd(a + 3, coef * v.w);
}

extern "C" void kernel_launch(void* const* d_in, const int* in_sizes, int n_in,
                              void* d_out, int out_size, void* d_ws, size_t ws_size,
                              hipStream_t stream) {
    const float* x      = (const float*)d_in[0];
    const int*   ei     = (const int*)  d_in[1];
    const float* W1     = (const float*)d_in[2];
    const float* a_src1 = (const float*)d_in[3];
    const float* a_dst1 = (const float*)d_in[4];
    const float* b1     = (const float*)d_in[5];
    const float* W2     = (const float*)d_in[6];
    const float* a_src2 = (const float*)d_in[7];
    const float* a_dst2 = (const float*)d_in[8];
    const float* b2     = (const float*)d_in[9];
    float* out = (float*)d_out;

    const int N  = in_sizes[0] / 128;   // 50000
    const int E  = in_sizes[1] / 2;     // 600000
    const int EA = E + N;               // with self loops
    const int* src = ei;
    const int* dst = ei + E;

    // ---- workspace layout ----
    char* w = (char*)d_ws;
    size_t off = 0;
    auto alloc = [&](size_t bytes) { size_t o = off; off = (off + bytes + 255) & ~(size_t)255; return o; };
    _Float16* Ah   = (_Float16*)(w + alloc((size_t)N * 128 * 2)); // layer1 A f16; reused as layer2 A
    _Float16* Bt1  = (_Float16*)(w + alloc(128 * 128 * 2));
    _Float16* Bt2  = (_Float16*)(w + alloc(64 * 128 * 2));
    float*    H1   = (float*)   (w + alloc((size_t)N * 128 * 4)); // gemm1 out [N,2,64]
    float*    ACC1 = (float*)   (w + alloc((size_t)N * 128 * 4));
    float*    AS1  = (float*)   (w + alloc((size_t)N * 2 * 4));
    float*    AD1  = (float*)   (w + alloc((size_t)N * 2 * 4));
    unsigned* M1   = (unsigned*)(w + alloc((size_t)N * 2 * 4));
    float*    DEN1 = (float*)   (w + alloc((size_t)N * 2 * 4));
    float*    EX1  = (float*)   (w + alloc((size_t)EA * 2 * 4));
    float*    H2   = (float*)   (w + alloc((size_t)N * 64 * 4));  // gemm2 out [N,1,64]
    float*    AS2  = (float*)   (w + alloc((size_t)N * 4));
    float*    AD2  = (float*)   (w + alloc((size_t)N * 4));
    unsigned* M2   = (unsigned*)(w + alloc((size_t)N * 4));
    float*    DEN2 = (float*)   (w + alloc((size_t)N * 4));
    float*    EX2  = (float*)   (w + alloc((size_t)EA * 4));

    auto cdiv = [](long long a, long long b) { return (int)((a + b - 1) / b); };
    const int B = 256;

    // ---- layer 1 ----
    k_f32_to_f16<<<cdiv((long long)N * 128, B), B, 0, stream>>>(x, Ah, N * 128);
    k_transpose_to_f16<<<cdiv(128 * 128, B), B, 0, stream>>>(W1, Bt1, 128, 128);
    k_transpose_to_f16<<<cdiv(128 * 64, B), B, 0, stream>>>(W2, Bt2, 128, 64);

    k_wmma_gemm<<<N / 16, 256, 0, stream>>>(Ah, Bt1, H1, 128);              // h1 = x@W1

    k_alphas<2, 64><<<cdiv((long long)N * 2, B), B, 0, stream>>>(H1, a_src1, a_dst1, AS1, AD1, N);
    k_fill_u32<<<cdiv((long long)N * 2, B), B, 0, stream>>>(M1, 0u, N * 2);
    k_fill_f32<<<cdiv((long long)N * 2, B), B, 0, stream>>>(DEN1, 0.f, N * 2);
    k_fill_f32<<<cdiv((long long)N * 128, B), B, 0, stream>>>(ACC1, 0.f, N * 128);

    k_edge_max<2><<<cdiv((long long)EA * 2, B), B, 0, stream>>>(src, dst, AS1, AD1, M1, EX1, E, EA);
    k_edge_exp<2><<<cdiv((long long)EA * 2, B), B, 0, stream>>>(dst, M1, EX1, DEN1, E, EA);
    k_edge_scatter<2, 64><<<cdiv((long long)EA * 32, B), B, 0, stream>>>(src, dst, H1, EX1, DEN1, ACC1, E, EA);

    // elu(acc + b1) -> f16 input of layer 2 (reuse Ah)
    k_bias_elu_f16<<<cdiv((long long)N * 128, B), B, 0, stream>>>(ACC1, b1, Ah, N * 128, 127);

    // ---- layer 2 ----
    k_wmma_gemm<<<N / 16, 128, 0, stream>>>(Ah, Bt2, H2, 64);               // h2 = elu@W2

    k_alphas<1, 64><<<cdiv((long long)N, B), B, 0, stream>>>(H2, a_src2, a_dst2, AS2, AD2, N);
    k_fill_u32<<<cdiv((long long)N, B), B, 0, stream>>>(M2, 0u, N);
    k_fill_f32<<<cdiv((long long)N, B), B, 0, stream>>>(DEN2, 0.f, N);
    k_bias_init<<<cdiv((long long)N * 64, B), B, 0, stream>>>(out, b2, N * 64, 63);

    k_edge_max<1><<<cdiv((long long)EA, B), B, 0, stream>>>(src, dst, AS2, AD2, M2, EX2, E, EA);
    k_edge_exp<1><<<cdiv((long long)EA, B), B, 0, stream>>>(dst, M2, EX2, DEN2, E, EA);
    k_edge_scatter<1, 64><<<cdiv((long long)EA * 16, B), B, 0, stream>>>(src, dst, H2, EX2, DEN2, out, E, EA);
}
